// GlimpseModel_46806553592052
// MI455X (gfx1250) — matrix operand order repaired and checked by reading, not compiled, not run on hardware
//
#include <hip/hip_runtime.h>

// Problem constants (match reference)
#define BATCH 128
#define KDIM  144
#define HDIM  512
#define WDIM  512
#define EPSV  1e-7f
#define NKT   9     // 144 / 16 k-tiles
#define WITER 16    // 512 / 32 w-steps (WMMA K=32)

typedef __attribute__((ext_vector_type(16))) __bf16 v16bf;
typedef __attribute__((ext_vector_type(8)))  float  v8f;

// linspace(-1, 1, 512) coordinate
__device__ __forceinline__ float gcoord(int i) {
    return -1.0f + (2.0f / 511.0f) * (float)i;
}

// ---------------------------------------------------------------------------
// Kernel 0: zero the raw accumulator in workspace
// ---------------------------------------------------------------------------
__global__ void glimpse_zero(float* __restrict__ raw, int n) {
    int i = blockIdx.x * blockDim.x + threadIdx.x;
    if (i < n) raw[i] = 0.0f;
}

// ---------------------------------------------------------------------------
// Kernel 1: fused bf16-WMMA pooling
//   grid = BATCH * 4 blocks (4 h-chunks of 128 rows), 256 threads = 8 waves.
//   Each wave: one 16-row h-tile, all 9 k-tiles accumulated in VGPRs.
//   raw[b,k] += sum_h ky[b,k,h] * (sum_w kx[b,k,w] * imgs[b,h,w])
// ---------------------------------------------------------------------------
__global__ __launch_bounds__(256, 1)
void glimpse_gemm(const float* __restrict__ imgs,
                  const float* __restrict__ s_c,
                  const float* __restrict__ s_z,
                  const float* __restrict__ mu,
                  const float* __restrict__ sigma,
                  float* __restrict__ raw) {
    // kx in WMMA A-fragment order: [kt][wit][lane][e], 147456 bytes of LDS
    __shared__ __attribute__((aligned(32))) __bf16 sKx[NKT * WITER * 32 * 16];

    const int b    = blockIdx.x >> 2;
    const int hc   = blockIdx.x & 3;
    const int tid  = threadIdx.x;
    const int lane = tid & 31;
    const int wave = tid >> 5;

    const float sc0 = s_c[2 * b + 0];
    const float sc1 = s_c[2 * b + 1];
    const float sz  = s_z[b];

    // ---- Precompute kx into LDS, already in A-fragment layout ----
    // idx = ((kt*16 + wit)*32 + ln)*16 + e
    for (int idx = tid; idx < NKT * WITER * 32 * 16; idx += 256) {
        const int e    = idx & 15;
        const int ln   = (idx >> 4) & 31;
        const int wit  = (idx >> 9) & 15;
        const int kt   = idx >> 13;
        const int k    = kt * 16 + (ln & 15);             // A-matrix row M
        const int half = ln >> 4;
        const int kk   = e + 16 * half;  // A-matrix col K
        const int w    = wit * 32 + kk;
        const float mux = (sc0 + mu[2 * k]) * sz;
        const float sg  = sigma[k] * sz;
        const float inv = -0.5f / (sg * sg);
        const float d   = gcoord(w) - mux;
        sKx[idx] = (__bf16)__expf(inv * d * d);
    }
    __syncthreads();

    const int half = lane >> 4;
    const int n    = lane & 15;                      // C/D column, B column
    const int h    = hc * 128 + wave * 16 + n;       // this lane's image row
    const float* imrow = imgs + ((size_t)b * HDIM + h) * (size_t)WDIM;

    v8f acc[NKT];
#pragma unroll
    for (int kt = 0; kt < NKT; ++kt)
        acc[kt] = v8f{0.f, 0.f, 0.f, 0.f, 0.f, 0.f, 0.f, 0.f};

    // ---- Main loop over W in steps of 32 (WMMA contraction) ----
    for (int wit = 0; wit < WITER; ++wit) {
        // B fragment source: 64B contiguous per lane (K = e + 16*half)
        const float4* p = (const float4*)(imrow + wit * 32 + half * 16);
        float4 f0 = p[0], f1 = p[1], f2 = p[2], f3 = p[3];
        if (wit + 1 < WITER)
            __builtin_prefetch(imrow + (wit + 1) * 32 + half * 16, 0, 0);

        // Issue ALL A-fragment LDS loads up front so the 18 ds_load_b128
        // drain under the WMMA chain instead of serializing per tile.
        v16bf af[NKT];
#pragma unroll
        for (int kt = 0; kt < NKT; ++kt)
            af[kt] = *(const v16bf*)&sKx[(((kt * WITER) + wit) * 32 + lane) * 16];

        v16bf bf;
        bf[0]  = (__bf16)f0.x; bf[1]  = (__bf16)f0.y; bf[2]  = (__bf16)f0.z; bf[3]  = (__bf16)f0.w;
        bf[4]  = (__bf16)f1.x; bf[5]  = (__bf16)f1.y; bf[6]  = (__bf16)f1.z; bf[7]  = (__bf16)f1.w;
        bf[8]  = (__bf16)f2.x; bf[9]  = (__bf16)f2.y; bf[10] = (__bf16)f2.z; bf[11] = (__bf16)f2.w;
        bf[12] = (__bf16)f3.x; bf[13] = (__bf16)f3.y; bf[14] = (__bf16)f3.z; bf[15] = (__bf16)f3.w;

        // Same B operand across the 9-tile chain -> reuse_b hint on all but
        // the last WMMA. Hints must be LITERAL constants (frontend check),
        // so unroll manually via macro.
#define WMMA_STEP(KT, RB)                                                     \
        acc[KT] = __builtin_amdgcn_wmma_f32_16x16x32_bf16(                    \
            false, af[KT], false, bf, (short)0, acc[KT],                      \
            /*reuse_a=*/false, /*reuse_b=*/RB)
        WMMA_STEP(0, true);
        WMMA_STEP(1, true);
        WMMA_STEP(2, true);
        WMMA_STEP(3, true);
        WMMA_STEP(4, true);
        WMMA_STEP(5, true);
        WMMA_STEP(6, true);
        WMMA_STEP(7, true);
        WMMA_STEP(8, false);
#undef WMMA_STEP
    }

    // ---- Fold in ky, reduce over the 16 h-columns, accumulate raw ----
    const float gyh = gcoord(h);
#pragma unroll
    for (int kt = 0; kt < NKT; ++kt) {
#pragma unroll
        for (int r = 0; r < 8; ++r) {
            const int k = kt * 16 + r + 8 * half;    // C row M
            const float muy = (sc1 + mu[2 * k + 1]) * sz;
            const float sg  = sigma[k] * sz;
            const float inv = -0.5f / (sg * sg);
            const float d   = gyh - muy;
            float partial = acc[kt][r] * __expf(inv * d * d);
            // butterfly reduce within each 16-lane half-group
            partial += __shfl_xor(partial, 1, 32);
            partial += __shfl_xor(partial, 2, 32);
            partial += __shfl_xor(partial, 4, 32);
            partial += __shfl_xor(partial, 8, 32);
            if (n == 0)
                atomicAdd(&raw[b * KDIM + k], partial);
        }
    }
}

// ---------------------------------------------------------------------------
// Kernel 2: norm = (sum_h ky)*(sum_w kx) + eps ; out = raw / norm
//   one wave per (b,k) pair
// ---------------------------------------------------------------------------
__global__ __launch_bounds__(256)
void glimpse_norm(const float* __restrict__ s_c,
                  const float* __restrict__ s_z,
                  const float* __restrict__ mu,
                  const float* __restrict__ sigma,
                  const float* __restrict__ raw,
                  float* __restrict__ out) {
    const int gw   = (blockIdx.x * 256 + threadIdx.x) >> 5;  // global wave id
    const int lane = threadIdx.x & 31;
    if (gw >= BATCH * KDIM) return;
    const int b = gw / KDIM;
    const int k = gw % KDIM;

    const float sc0 = s_c[2 * b + 0];
    const float sc1 = s_c[2 * b + 1];
    const float sz  = s_z[b];
    const float mux = (sc0 + mu[2 * k + 0]) * sz;
    const float muy = (sc1 + mu[2 * k + 1]) * sz;
    const float sg  = sigma[k] * sz;
    const float inv = -0.5f / (sg * sg);

    float sx = 0.0f, sy = 0.0f;
    for (int i = lane; i < WDIM; i += 32) {
        const float d = gcoord(i) - mux;
        sx += __expf(inv * d * d);
    }
    for (int i = lane; i < HDIM; i += 32) {
        const float d = gcoord(i) - muy;
        sy += __expf(inv * d * d);
    }
    for (int m = 1; m < 32; m <<= 1) {
        sx += __shfl_xor(sx, m, 32);
        sy += __shfl_xor(sy, m, 32);
    }
    if (lane == 0)
        out[gw] = raw[gw] / (sy * sx + EPSV);
}

// ---------------------------------------------------------------------------
extern "C" void kernel_launch(void* const* d_in, const int* in_sizes, int n_in,
                              void* d_out, int out_size, void* d_ws, size_t ws_size,
                              hipStream_t stream) {
    const float* imgs  = (const float*)d_in[0];   // (B,H,W)
    const float* s_c   = (const float*)d_in[1];   // (B,2)
    const float* s_z   = (const float*)d_in[2];   // (B,1)
    const float* mu    = (const float*)d_in[3];   // (K,2)
    const float* sigma = (const float*)d_in[4];   // (K,)
    float* out = (float*)d_out;                   // (B,K)
    float* raw = (float*)d_ws;                    // B*K f32 accumulator

    const int nbk = BATCH * KDIM;
    glimpse_zero<<<(nbk + 255) / 256, 256, 0, stream>>>(raw, nbk);

    // 4 h-chunks of 128 rows per batch; 8 waves/block, 1 h-tile each, all 9 k-tiles
    glimpse_gemm<<<BATCH * 4, 256, 0, stream>>>(imgs, s_c, s_z, mu, sigma, raw);

    // one wave32 per (b,k): 18432 waves -> 2304 blocks of 8 waves
    glimpse_norm<<<(nbk + 7) / 8, 256, 0, stream>>>(s_c, s_z, mu, sigma, raw, out);
}
